// SLSTM_block_78778290143630
// MI455X (gfx1250) — compile-verified, matrix operand order checked by reading, not run on hardware
//
#include <hip/hip_runtime.h>
#include <hip/hip_bf16.h>

// Problem constants (match reference)
#define B_   8
#define S_   1024
#define H_   512
#define H2_  1024
#define H4_  2048
#define H7_  3584
#define BS_  (B_ * S_)
#define L_   4

typedef __bf16 bf16;
typedef __attribute__((ext_vector_type(16))) __bf16 v16bf;
typedef __attribute__((ext_vector_type(8)))  __bf16 v8bf;
typedef __attribute__((ext_vector_type(8)))  float  v8f;

__device__ __forceinline__ float sigmoidf_(float x) { return 1.f / (1.f + expf(-x)); }

// ---------------- block reduction (256 threads = 8 waves, wave32) -----------
__device__ __forceinline__ float blockReduceSum256(float v, float* sh) {
#pragma unroll
  for (int off = 16; off > 0; off >>= 1) v += __shfl_down(v, off, 32);
  int t = threadIdx.x;
  if ((t & 31) == 0) sh[t >> 5] = v;
  __syncthreads();
  float r = (t < 8) ? sh[t] : 0.f;
  if (t < 32) {
#pragma unroll
    for (int off = 4; off > 0; off >>= 1) r += __shfl_down(r, off, 32);
    if (t == 0) sh[0] = r;
  }
  __syncthreads();
  r = sh[0];
  __syncthreads();
  return r;
}

// ---------------- bf16 WMMA GEMM: C[M,N] = A[M,K] * W[N,K]^T (+bias +addend)
// A fragment layout per CDNA5 ISA 7.12.2 (16-bit A 16x32):
//   lane = M row (lane&15), lane-half selects K {0..7,16..23} vs {8..15,24..31}.
// B operand = rows of W (D = A x B with B = W^T), mirrored layout (lane = N col).
__device__ __forceinline__ v16bf load_frag(const bf16* __restrict__ base, int ld,
                                           int k, int lane) {
  int m  = lane & 15;
  int kb = (lane >> 4) << 3;  // 0 or 8
  const bf16* p = base + (size_t)m * ld + k + kb;
  v8bf lo = *(const v8bf*)(p);       // K = kb .. kb+7        (16B load)
  v8bf hi = *(const v8bf*)(p + 16);  // K = kb+16 .. kb+23    (16B load)
  return __builtin_shufflevector(lo, hi, 0, 1, 2, 3, 4, 5, 6, 7,
                                 8, 9, 10, 11, 12, 13, 14, 15);
}

template <bool HAS_BIAS, bool HAS_ADDEND>
__global__ __launch_bounds__(128) void gemm_bf16_nt(
    const bf16* __restrict__ A, int lda,
    const bf16* __restrict__ W, int ldb,
    float* __restrict__ C, int ldc, int K,
    const float* __restrict__ bias,     // len N (HAS_BIAS)
    const float* __restrict__ addend) { // M x ldc (HAS_ADDEND)
  const int wave = threadIdx.x >> 5;
  const int lane = threadIdx.x & 31;
  const int m0 = blockIdx.x * 128 + (wave >> 1) * 64;
  const int n0 = blockIdx.y * 128 + (wave & 1) * 64;

  v8f acc[4][4];
  v8f zero = 0.f;
#pragma unroll
  for (int i = 0; i < 4; ++i)
#pragma unroll
    for (int j = 0; j < 4; ++j) acc[i][j] = zero;

  for (int k = 0; k < K; k += 32) {
    v16bf af[4], bfr[4];
#pragma unroll
    for (int i = 0; i < 4; ++i)
      af[i] = load_frag(A + (size_t)(m0 + 16 * i) * lda, lda, k, lane);
#pragma unroll
    for (int j = 0; j < 4; ++j)
      bfr[j] = load_frag(W + (size_t)(n0 + 16 * j) * ldb, ldb, k, lane);
#pragma unroll
    for (int i = 0; i < 4; ++i)
#pragma unroll
      for (int j = 0; j < 4; ++j)
        acc[i][j] = __builtin_amdgcn_wmma_f32_16x16x32_bf16(
            false, af[i], false, bfr[j], (short)0, acc[i][j], false, false);
  }

  // C/D layout: VGPR v -> M = v (lanes 0-15) / v+8 (lanes 16-31), N = lane&15
  const int nl = lane & 15;
  const int mo = (lane >> 4) << 3;
#pragma unroll
  for (int i = 0; i < 4; ++i) {
    const size_t rbase = (size_t)(m0 + 16 * i + mo) * ldc;
    float* __restrict__ crow = C + rbase;
    const float* __restrict__ arow = HAS_ADDEND ? (addend + rbase) : nullptr;
#pragma unroll
    for (int j = 0; j < 4; ++j) {
      const int col = n0 + 16 * j + nl;
      float bv = HAS_BIAS ? bias[col] : 0.f;
#pragma unroll
      for (int v = 0; v < 8; ++v) {
        float o = acc[i][j][v] + bv;
        if (HAS_ADDEND) o += arow[(size_t)v * ldc + col];
        crow[(size_t)v * ldc + col] = o;
      }
    }
  }
}

// ---------------- small elementwise / reduction kernels ----------------------
__global__ void cvt_kernel(const float* __restrict__ in, bf16* __restrict__ out, int n) {
  int i = blockIdx.x * 256 + threadIdx.x;
  if (i < n) out[i] = (bf16)in[i];
}

__global__ void lengths_kernel(const float* __restrict__ mask, float* __restrict__ len) {
  __shared__ float sh[8];
  int b = blockIdx.x;
  float acc = 0.f;
  for (int s = threadIdx.x; s < S_; s += 256) acc += 1.f - mask[b * S_ + s];
  acc = blockReduceSum256(acc, sh);
  if (threadIdx.x == 0) len[b] = acc;
}

// out[b,h] = sum_s X[b,s,h] / len[b]   (512 threads: one per h, coalesced)
__global__ void colsum_kernel(const float* __restrict__ X, const float* __restrict__ len,
                              float* __restrict__ out1, float* __restrict__ out2) {
  int b = blockIdx.x, h = threadIdx.x;
  float acc = 0.f;
  for (int s = 0; s < S_; ++s) acc += X[((size_t)b * S_ + s) * H_ + h];
  float v = acc / len[b];
  out1[b * H_ + h] = v;
  if (out2) out2[b * H_ + h] = v;
}

// Fused: K=3 shift-window sums for h and c, masking, bf16 pack of the
// [h, hb, ha, g_h_exp] GEMM operand; c_before/after kept in f32 for cell update.
__global__ void buildA_kernel(const float* __restrict__ wh, const float* __restrict__ wc,
                              const float* __restrict__ dgh, const float* __restrict__ mask,
                              bf16* __restrict__ A, float* __restrict__ cb,
                              float* __restrict__ ca) {
  int idx = blockIdx.x * 256 + threadIdx.x;  // BS_*H_ total
  int h = idx & (H_ - 1);
  int r = idx >> 9;
  int s = r & (S_ - 1);
  int b = r >> 10;
  float hb = 0.f, ha = 0.f, cbv = 0.f, cav = 0.f;
#pragma unroll
  for (int d = 1; d <= 3; ++d) {
    if (s - d >= 0) { hb += wh[(size_t)(r - d) * H_ + h]; cbv += wc[(size_t)(r - d) * H_ + h]; }
    if (s + d < S_) { ha += wh[(size_t)(r + d) * H_ + h]; cav += wc[(size_t)(r + d) * H_ + h]; }
  }
  float sm = 1.f - mask[b * S_ + s];
  size_t arow = (size_t)r * H4_;
  A[arow + h]           = (bf16)wh[(size_t)r * H_ + h];
  A[arow + H_ + h]      = (bf16)(hb * sm);
  A[arow + 2 * H_ + h]  = (bf16)(ha * sm);
  A[arow + 3 * H_ + h]  = (bf16)(dgh[b * H_ + h] * sm);
  cb[(size_t)r * H_ + h] = cbv * sm;
  ca[(size_t)r * H_ + h] = cav * sm;
}

// Fused: 7 layernorms per row (values stay in registers: each thread owns
// h = t and h = t+256), sigmoid + 5-way gate softmax, cell/hidden update, mask.
__global__ __launch_bounds__(256) void gate_kernel(
    const float* __restrict__ gates, const float* __restrict__ cb,
    const float* __restrict__ ca, float* __restrict__ wh, float* __restrict__ wc,
    const float* __restrict__ dgc, const float* __restrict__ ln_g,
    const float* __restrict__ ln_b, const float* __restrict__ mask) {
  __shared__ float sh[8];
  int r = blockIdx.x;
  int b = r >> 10;
  int s = r & (S_ - 1);
  const float* grow = gates + (size_t)r * H7_;
  float vloc[7][2];
#pragma unroll
  for (int i = 0; i < 7; ++i) {
    float x0 = grow[i * H_ + threadIdx.x];
    float x1 = grow[i * H_ + threadIdx.x + 256];
    float s1 = blockReduceSum256(x0 + x1, sh);
    float s2 = blockReduceSum256(x0 * x0 + x1 * x1, sh);
    float mean = s1 * (1.f / H_);
    float var  = s2 * (1.f / H_) - mean * mean;
    float rstd = rsqrtf(var + 1e-5f);
    int h0 = threadIdx.x, h1 = threadIdx.x + 256;
    vloc[i][0] = (x0 - mean) * rstd * ln_g[i * H_ + h0] + ln_b[i * H_ + h0];
    vloc[i][1] = (x1 - mean) * rstd * ln_g[i * H_ + h1] + ln_b[i * H_ + h1];
  }
  float sm = 1.f - mask[b * S_ + s];
#pragma unroll
  for (int j = 0; j < 2; ++j) {
    int h = threadIdx.x + 256 * j;
    float e[5];
    float mx = -1e30f;
#pragma unroll
    for (int g = 0; g < 5; ++g) { e[g] = sigmoidf_(vloc[g][j]); mx = fmaxf(mx, e[g]); }
    float Z = 0.f;
#pragma unroll
    for (int g = 0; g < 5; ++g) { e[g] = expf(e[g] - mx); Z += e[g]; }
    float inv = 1.f / Z;
    size_t idx = (size_t)r * H_ + h;
    float cnew = tanhf(vloc[6][j]) * e[0] * inv + cb[idx] * e[1] * inv +
                 ca[idx] * e[2] * inv + wc[idx] * e[3] * inv +
                 dgc[b * H_ + h] * e[4] * inv;
    float hnew = sigmoidf_(vloc[5][j]) * tanhf(cnew);
    wc[idx] = cnew * sm;
    wh[idx] = hnew * sm;
  }
}

// fo = [dgh, wavg] @ fo_w^T + fo_b  (tiny: 8x1024x1024)
__global__ __launch_bounds__(256) void fo_gemm_kernel(
    const float* __restrict__ dgh, const float* __restrict__ wavg,
    const float* __restrict__ fo_w, const float* __restrict__ fo_b,
    float* __restrict__ fo) {
  __shared__ float catv[H2_];
  int b = blockIdx.x;
  for (int k = threadIdx.x; k < H2_; k += 256)
    catv[k] = (k < H_) ? dgh[b * H_ + k] : wavg[b * H_ + k - H_];
  __syncthreads();
  for (int n = threadIdx.x; n < H2_; n += 256) {
    float acc = fo_b[n];
    const float* wrow = fo_w + (size_t)n * H2_;
    for (int k = 0; k < H2_; ++k) acc += catv[k] * wrow[k];
    fo[b * H2_ + n] = acc;
  }
}

// d_f / d_o = layernorm(sigmoid(fo half), ln_g[7+half])
__global__ __launch_bounds__(256) void dfo_kernel(
    const float* __restrict__ fo, float* __restrict__ d_f, float* __restrict__ d_o,
    const float* __restrict__ ln_g, const float* __restrict__ ln_b) {
  __shared__ float sh[8];
  int b = blockIdx.x >> 1, half = blockIdx.x & 1;
  const float* row = fo + b * H2_ + half * H_;
  float x0 = sigmoidf_(row[threadIdx.x]);
  float x1 = sigmoidf_(row[threadIdx.x + 256]);
  float s1 = blockReduceSum256(x0 + x1, sh);
  float s2 = blockReduceSum256(x0 * x0 + x1 * x1, sh);
  float mean = s1 * (1.f / H_);
  float var  = s2 * (1.f / H_) - mean * mean;
  float rstd = rsqrtf(var + 1e-5f);
  int h0 = threadIdx.x, h1 = threadIdx.x + 256;
  int g = (7 + half) * H_;
  float* out = half ? d_o : d_f;
  out[b * H_ + h0] = (x0 - mean) * rstd * ln_g[g + h0] + ln_b[g + h0];
  out[b * H_ + h1] = (x1 - mean) * rstd * ln_g[g + h1] + ln_b[g + h1];
}

__global__ void buildA2_kernel(const float* __restrict__ wh, const float* __restrict__ dgh,
                               const float* __restrict__ mask, bf16* __restrict__ A2) {
  int idx = blockIdx.x * 256 + threadIdx.x;  // BS_*H2_ total
  int h2 = idx & (H2_ - 1);
  int r = idx >> 10;
  int s = r & (S_ - 1);
  int b = r >> 10;
  float v;
  if (h2 < H_) v = wh[(size_t)r * H_ + h2];
  else         v = dgh[b * H_ + h2 - H_] * (1.f - mask[b * S_ + s]);
  A2[(size_t)r * H2_ + h2] = (bf16)v;
}

// In-place: f1 <- sigmoid(layernorm(f1, ln_g[9]))
__global__ __launch_bounds__(256) void f1post_kernel(float* __restrict__ f1,
                                                     const float* __restrict__ ln_g,
                                                     const float* __restrict__ ln_b) {
  __shared__ float sh[8];
  int r = blockIdx.x;
  float* row = f1 + (size_t)r * H_;
  float x0 = row[threadIdx.x], x1 = row[threadIdx.x + 256];
  float s1 = blockReduceSum256(x0 + x1, sh);
  float s2 = blockReduceSum256(x0 * x0 + x1 * x1, sh);
  float mean = s1 * (1.f / H_);
  float var  = s2 * (1.f / H_) - mean * mean;
  float rstd = rsqrtf(var + 1e-5f);
  int h0 = threadIdx.x, h1 = threadIdx.x + 256;
  row[h0] = sigmoidf_((x0 - mean) * rstd * ln_g[9 * H_ + h0] + ln_b[9 * H_ + h0]);
  row[h1] = sigmoidf_((x1 - mean) * rstd * ln_g[9 * H_ + h1] + ln_b[9 * H_ + h1]);
}

// Softmax over the (S+1) length axis per (b,h), fused with the
// sum(g_f * word_c) contraction; updates dummy_g_c / dummy_g_h.
__global__ void seqsm_kernel(const float* __restrict__ f1, const float* __restrict__ mask,
                             const float* __restrict__ d_f, const float* __restrict__ d_o,
                             const float* __restrict__ wc, float* __restrict__ dgc,
                             float* __restrict__ dgh) {
  int b = blockIdx.x, h = threadIdx.x;
  float df = d_f[b * H_ + h];
  float m = df;
  for (int s = 0; s < S_; ++s) {
    float x = f1[((size_t)b * S_ + s) * H_ + h] - mask[b * S_ + s] * 1e25f;
    m = fmaxf(m, x);
  }
  float el = expf(df - m);
  float Z = el;
  float W = el * dgc[b * H_ + h];
  for (int s = 0; s < S_; ++s) {
    size_t idx = ((size_t)b * S_ + s) * H_ + h;
    float x = f1[idx] - mask[b * S_ + s] * 1e25f;
    float e = expf(x - m);
    Z += e;
    W += e * wc[idx];
  }
  float c = W / Z;
  dgc[b * H_ + h] = c;
  dgh[b * H_ + h] = d_o[b * H_ + h] * tanhf(c);
}

// ---------------------------------------------------------------------------
extern "C" void kernel_launch(void* const* d_in, const int* in_sizes, int n_in,
                              void* d_out, int out_size, void* d_ws, size_t ws_size,
                              hipStream_t stream) {
  const float* word_h = (const float*)d_in[0];
  const float* mask   = (const float*)d_in[1];
  /* src_tokens d_in[2] unused */
  const float* WV_w = (const float*)d_in[3];
  const float* WV_b = (const float*)d_in[4];
  const float* U_w  = (const float*)d_in[5];
  const float* fo_w = (const float*)d_in[6];
  const float* fo_b = (const float*)d_in[7];
  const float* f1_w = (const float*)d_in[8];
  const float* f1_b = (const float*)d_in[9];
  const float* ln_g = (const float*)d_in[10];
  const float* ln_b = (const float*)d_in[11];

  char* base = (char*)d_ws;
  size_t off = 0;
  auto carve = [&](size_t bytes) -> void* {
    void* p = base + off;
    off += (bytes + 255) & ~(size_t)255;
    return p;
  };
  bf16*  wv_bf  = (bf16*)carve((size_t)H7_ * H4_ * 2);
  bf16*  uw_bf  = (bf16*)carve((size_t)H7_ * H_ * 2);
  bf16*  f1w_bf = (bf16*)carve((size_t)H_ * H2_ * 2);
  bf16*  hbf    = (bf16*)carve((size_t)BS_ * H_ * 2);
  bf16*  Abuf   = (bf16*)carve((size_t)BS_ * H4_ * 2);  // reused for A2
  float* Ux     = (float*)carve((size_t)BS_ * H7_ * 4);
  float* gates  = (float*)carve((size_t)BS_ * H7_ * 4);
  float* wh     = (float*)carve((size_t)BS_ * H_ * 4);
  float* wc     = (float*)carve((size_t)BS_ * H_ * 4);
  float* cb     = (float*)carve((size_t)BS_ * H_ * 4);
  float* ca     = (float*)carve((size_t)BS_ * H_ * 4);
  float* f1out  = (float*)carve((size_t)BS_ * H_ * 4);
  float* dgh    = (float*)carve((size_t)B_ * H_ * 4);
  float* dgc    = (float*)carve((size_t)B_ * H_ * 4);
  float* wavg   = (float*)carve((size_t)B_ * H_ * 4);
  float* d_f    = (float*)carve((size_t)B_ * H_ * 4);
  float* d_o    = (float*)carve((size_t)B_ * H_ * 4);
  float* fo     = (float*)carve((size_t)B_ * H2_ * 4);
  float* len    = (float*)carve((size_t)B_ * 4);

  // ---- one-time prep ----
  lengths_kernel<<<B_, 256, 0, stream>>>(mask, len);
  cvt_kernel<<<(H7_ * H4_ + 255) / 256, 256, 0, stream>>>(WV_w, wv_bf, H7_ * H4_);
  cvt_kernel<<<(H7_ * H_ + 255) / 256, 256, 0, stream>>>(U_w, uw_bf, H7_ * H_);
  cvt_kernel<<<(H_ * H2_ + 255) / 256, 256, 0, stream>>>(f1_w, f1w_bf, H_ * H2_);
  hipMemcpyAsync(wh, word_h, (size_t)BS_ * H_ * 4, hipMemcpyDeviceToDevice, stream);
  hipMemcpyAsync(wc, word_h, (size_t)BS_ * H_ * 4, hipMemcpyDeviceToDevice, stream);
  cvt_kernel<<<(BS_ * H_ + 255) / 256, 256, 0, stream>>>(word_h, hbf, BS_ * H_);
  colsum_kernel<<<B_, H_, 0, stream>>>(wh, len, dgh, dgc);

  // Ux = word_h @ U_w^T   (8192 x 3584, K=512), no bias/addend
  gemm_bf16_nt<false, false><<<dim3(BS_ / 128, H7_ / 128), 128, 0, stream>>>(
      hbf, H_, uw_bf, H_, Ux, H7_, H_, nullptr, nullptr);

  // ---- L layers ----
  for (int l = 0; l < L_; ++l) {
    buildA_kernel<<<BS_ * H_ / 256, 256, 0, stream>>>(wh, wc, dgh, mask, Abuf, cb, ca);
    // all_gates = A @ WV_w^T + WV_b + Ux   (8192 x 3584, K=2048)
    gemm_bf16_nt<true, true><<<dim3(BS_ / 128, H7_ / 128), 128, 0, stream>>>(
        Abuf, H4_, wv_bf, H4_, gates, H7_, H4_, WV_b, Ux);
    gate_kernel<<<BS_, 256, 0, stream>>>(gates, cb, ca, wh, wc, dgc, ln_g, ln_b, mask);
    colsum_kernel<<<B_, H_, 0, stream>>>(wh, len, wavg, nullptr);
    fo_gemm_kernel<<<B_, 256, 0, stream>>>(dgh, wavg, fo_w, fo_b, fo);
    dfo_kernel<<<2 * B_, 256, 0, stream>>>(fo, d_f, d_o, ln_g, ln_b);
    buildA2_kernel<<<BS_ * H2_ / 256, 256, 0, stream>>>(wh, dgh, mask, Abuf);
    // f1 = A2 @ f1_w^T + f1_b   (8192 x 512, K=1024)
    gemm_bf16_nt<true, false><<<dim3(BS_ / 128, H_ / 128), 128, 0, stream>>>(
        Abuf, H2_, f1w_bf, H2_, f1out, H_, H2_, f1_b, nullptr);
    f1post_kernel<<<BS_, 256, 0, stream>>>(f1out, ln_g, ln_b);
    seqsm_kernel<<<B_, H_, 0, stream>>>(f1out, mask, d_f, d_o, wc, dgc, dgh);
  }

  // ---- outputs: word_h, word_c, dummy_g_h, dummy_g_c (concat, f32) ----
  float* out = (float*)d_out;
  hipMemcpyAsync(out, wh, (size_t)BS_ * H_ * 4, hipMemcpyDeviceToDevice, stream);
  hipMemcpyAsync(out + (size_t)BS_ * H_, wc, (size_t)BS_ * H_ * 4,
                 hipMemcpyDeviceToDevice, stream);
  hipMemcpyAsync(out + 2 * (size_t)BS_ * H_, dgh, (size_t)B_ * H_ * 4,
                 hipMemcpyDeviceToDevice, stream);
  hipMemcpyAsync(out + 2 * (size_t)BS_ * H_ + (size_t)B_ * H_, dgc,
                 (size_t)B_ * H_ * 4, hipMemcpyDeviceToDevice, stream);
  (void)in_sizes; (void)n_in; (void)out_size; (void)ws_size;
}
